// CrossAttention_55130200211990
// MI455X (gfx1250) — matrix operand order repaired and checked by reading, not compiled, not run on hardware
//
#include <hip/hip_runtime.h>
#include <hip/hip_bf16.h>
#include <stdint.h>

// ---------------------------------------------------------------------------
// CrossAttention (B=4, C=384, H=W=64, nd=12 -> hd=32, n_head=1) for gfx1250.
//   1) repack_w: Wout (384,32,3,3) f32 -> Wt (9,384,32) bf16
//   2) proj_q:   q  = (Wq @ x) * scale  -> bf16 (B, 4096, 32)
//   3) proj_kv:  k  -> bf16 (B, 4096, 32), vT -> bf16 (B, 32, 4096)
//   4) attn:     flash attention, 64-key chunks:
//                  4x v_wmma S, stage-major packed-bf16 max reduce
//                  (ds_swizzle + v_pk_max_num_bf16), 4x v_wmma PV,
//                  2x v_wmma row-sum (P x ones -> denominator in every
//                  column, no cross-lane broadcast needed).
//                  Output bf16 position-major (B, 4096, 32).
//   5) conv3x3 as implicit GEMM: 9 accumulating v_wmma per 16co x 16x tile
//                  (K per tap = 32 ci = one WMMA K-step), fused bias +
//                  residual, coalesced f32 stores -> d_out (B,384,64,64).
// ---------------------------------------------------------------------------

typedef __bf16 bf16_t;
typedef __attribute__((ext_vector_type(16))) __bf16 v16bf;
typedef __attribute__((ext_vector_type(8)))  float  v8f;

#define NPOS 4096
#define HD   32
#define CIN  384
#define NB   4

union Frag16 { v16bf v; uint4 u[2]; };

__device__ __forceinline__ unsigned pack_bf16_trunc(float lo, float hi) {
    // truncation only lowers the max -> softmax shift-invariance keeps results exact
    return (__float_as_uint(hi) & 0xffff0000u) | (__float_as_uint(lo) >> 16);
}

__device__ __forceinline__ unsigned pk_max_bf16(unsigned a, unsigned b) {
    unsigned d;
    asm("v_pk_max_num_bf16 %0, %1, %2" : "=v"(d) : "v"(a), "v"(b));
    return d;
}

// ds_swizzle xor patterns, and_mask=0x1f (butterfly within 16-lane halves)
#define SWZ_XOR(k) ((k << 10) | 0x1f)

// ---------------------------------------------------------------------------
__global__ __launch_bounds__(256) void repack_w_kernel(
    const float* __restrict__ W,   // (384, 32, 3, 3)
    bf16_t* __restrict__ Wt)       // (9, 384, 32): [tap][co][ci]
{
    int g = blockIdx.x * 256 + threadIdx.x;   // 9*384*32 threads
    int ci  = g & 31;
    int co  = (g >> 5) % CIN;
    int tap = g / (CIN * HD);
    Wt[g] = (bf16_t)W[((size_t)co * HD + ci) * 9 + tap];
}

// ---------------------------------------------------------------------------
__global__ __launch_bounds__(256) void proj_q_kernel(
    const float* __restrict__ x, const float* __restrict__ Wq,
    bf16_t* __restrict__ q, float scale)
{
    int g  = blockIdx.x * 256 + threadIdx.x;      // NB*NPOS*HD threads
    int d  = g & 31;
    int p  = g >> 5;
    int hw = p & (NPOS - 1);
    int b  = p >> 12;
    const float* xp = x + (size_t)b * CIN * NPOS + hw;
    const float* wp = Wq + (size_t)d * CIN;
    float acc = 0.f;
    #pragma unroll 8
    for (int c = 0; c < CIN; ++c)
        acc = fmaf(wp[c], xp[(size_t)c * NPOS], acc);
    q[(size_t)p * HD + d] = (bf16_t)(acc * scale);
}

// ---------------------------------------------------------------------------
__global__ __launch_bounds__(256) void proj_kv_kernel(
    const float* __restrict__ xe, const float* __restrict__ Wkv,
    bf16_t* __restrict__ k, bf16_t* __restrict__ vT)
{
    int g  = blockIdx.x * 256 + threadIdx.x;      // NB*NPOS*64 threads
    int d  = g & 63;
    int p  = g >> 6;
    int hw = p & (NPOS - 1);
    int b  = p >> 12;
    const float* xp = xe + (size_t)b * CIN * NPOS + hw;
    const float* wp = Wkv + (size_t)d * CIN;
    float acc = 0.f;
    #pragma unroll 8
    for (int c = 0; c < CIN; ++c)
        acc = fmaf(wp[c], xp[(size_t)c * NPOS], acc);
    if (d < HD)
        k[((size_t)(b * NPOS + hw)) * HD + d] = (bf16_t)acc;
    else
        vT[((size_t)b * HD + (d - HD)) * NPOS + hw] = (bf16_t)acc;
}

// ---------------------------------------------------------------------------
// One wave = one 16-query tile; 4 waves per workgroup; 64-key chunks.
// ---------------------------------------------------------------------------
__global__ __launch_bounds__(128) void attn_kernel(
    const bf16_t* __restrict__ Q, const bf16_t* __restrict__ K,
    const bf16_t* __restrict__ VT, bf16_t* __restrict__ OP)
{
    __shared__ __align__(16) bf16_t pTile[4][16][64];   // per-wave 16x64 P tile

    const int lane = threadIdx.x & 31;
    const int wave = threadIdx.x >> 5;
    const int tile = blockIdx.x * 4 + wave;             // 0 .. NB*256-1
    const int b    = tile >> 8;
    const int q0   = (tile & 255) << 4;

    const int half = lane >> 4;
    const int l16  = lane & 15;

    // Q A-fragment (row m = l16, K per WMMA A layout)
    Frag16 aq;
    {
        const bf16_t* qp = Q + ((size_t)(b * NPOS + q0 + l16)) * HD + half * 8;
        aq.u[0] = *(const uint4*)(qp);
        aq.u[1] = *(const uint4*)(qp + 16);
    }

    // all-ones B fragment: C[m,n] = rowsum(A) replicated into every column
    Frag16 ones;
    #pragma unroll
    for (int i = 0; i < 16; ++i) ones.v[i] = (bf16_t)1.0f;

    float mrow[8];
    v8f o0 = {}, o1 = {}, csum = {};
    #pragma unroll
    for (int v = 0; v < 8; ++v) mrow[v] = -1e30f;

    const bf16_t* kbase = K  + (size_t)b * NPOS * HD;
    const bf16_t* vbase = VT + (size_t)b * HD * NPOS;
    const float LOG2E = 1.4426950408889634f;

    for (int j0 = 0; j0 < NPOS; j0 += 64) {
        // ---- 4 K B-fragments (16 keys each) + 4 S WMMAs ------------------
        v8f c[4];
        #pragma unroll
        for (int t = 0; t < 4; ++t) {
            Frag16 bk;
            const bf16_t* kp = kbase + (size_t)(j0 + t * 16 + l16) * HD + half * 8;
            bk.u[0] = *(const uint4*)(kp);
            bk.u[1] = *(const uint4*)(kp + 16);
            if (t == 0) __builtin_prefetch(kp + 64 * HD, 0, 3);
            v8f z = {};
            c[t] = __builtin_amdgcn_wmma_f32_16x16x32_bf16(false, aq.v, false, bk.v,
                                                           (short)0, z, false, false);
        }

        // ---- packed-bf16 row-max reduction over 64 keys, stage-major -----
        float mv[8];
        #pragma unroll
        for (int v = 0; v < 8; ++v)
            mv[v] = fmaxf(fmaxf(c[0][v], c[1][v]), fmaxf(c[2][v], c[3][v]));

        unsigned pk[4];
        #pragma unroll
        for (int j = 0; j < 4; ++j) pk[j] = pack_bf16_trunc(mv[2 * j], mv[2 * j + 1]);

        #define RED_STAGE(XK)                                                        \
        {                                                                            \
            unsigned t0 = (unsigned)__builtin_amdgcn_ds_swizzle((int)pk[0], SWZ_XOR(XK)); \
            unsigned t1 = (unsigned)__builtin_amdgcn_ds_swizzle((int)pk[1], SWZ_XOR(XK)); \
            unsigned t2 = (unsigned)__builtin_amdgcn_ds_swizzle((int)pk[2], SWZ_XOR(XK)); \
            unsigned t3 = (unsigned)__builtin_amdgcn_ds_swizzle((int)pk[3], SWZ_XOR(XK)); \
            pk[0] = pk_max_bf16(pk[0], t0);                                          \
            pk[1] = pk_max_bf16(pk[1], t1);                                          \
            pk[2] = pk_max_bf16(pk[2], t2);                                          \
            pk[3] = pk_max_bf16(pk[3], t3);                                          \
        }
        RED_STAGE(1) RED_STAGE(2) RED_STAGE(4) RED_STAGE(8)
        #undef RED_STAGE

        // ---- online rescale ---------------------------------------------
        #pragma unroll
        for (int v = 0; v < 8; ++v) {
            unsigned u = pk[v >> 1];
            float mx = (v & 1) ? __uint_as_float(u & 0xffff0000u)
                               : __uint_as_float(u << 16);
            float mnew = fmaxf(mrow[v], mx);
            float a = __builtin_amdgcn_exp2f((mrow[v] - mnew) * LOG2E);
            mrow[v] = mnew;
            o0[v] *= a;  o1[v] *= a;  csum[v] *= a;
        }

        // ---- P = exp(S - m), bf16, into per-wave LDS tile ----------------
        #pragma unroll
        for (int t = 0; t < 4; ++t)
            #pragma unroll
            for (int v = 0; v < 8; ++v) {
                float p = __builtin_amdgcn_exp2f((c[t][v] - mrow[v]) * LOG2E);
                pTile[wave][v + half * 8][l16 + 16 * t] = (bf16_t)p;
            }

        // ---- PV + row-sum WMMAs over the two 32-key sub-chunks -----------
        #pragma unroll
        for (int kc = 0; kc < 2; ++kc) {
            Frag16 ap;
            const bf16_t* pp = &pTile[wave][l16][kc * 32 + half * 8];
            ap.u[0] = *(const uint4*)(pp);
            ap.u[1] = *(const uint4*)(pp + 16);

            csum = __builtin_amdgcn_wmma_f32_16x16x32_bf16(false, ap.v, false, ones.v,
                                                           (short)0, csum, false, false);

            Frag16 bv0, bv1;
            const bf16_t* vp0 = vbase + (size_t)l16 * NPOS + j0 + kc * 32 + half * 8;
            bv0.u[0] = *(const uint4*)(vp0);
            bv0.u[1] = *(const uint4*)(vp0 + 16);
            const bf16_t* vp1 = vp0 + 16 * NPOS;
            bv1.u[0] = *(const uint4*)(vp1);
            bv1.u[1] = *(const uint4*)(vp1 + 16);
            if (kc == 0) __builtin_prefetch(vp0 + 64, 0, 3);

            o0 = __builtin_amdgcn_wmma_f32_16x16x32_bf16(false, ap.v, false, bv0.v,
                                                         (short)0, o0, false, false);
            o1 = __builtin_amdgcn_wmma_f32_16x16x32_bf16(false, ap.v, false, bv1.v,
                                                         (short)0, o1, false, false);
        }
    }

    // ---- normalize, write bf16 position-major (B, 4096, 32) --------------
    #pragma unroll
    for (int v = 0; v < 8; ++v) {
        float inv = 1.0f / csum[v];
        int qi = q0 + v + half * 8;
        bf16_t* op = OP + ((size_t)(b * NPOS) + qi) * HD;
        op[l16]      = (bf16_t)(o0[v] * inv);
        op[l16 + 16] = (bf16_t)(o1[v] * inv);
    }
}

// ---------------------------------------------------------------------------
// 3x3 conv as implicit GEMM. One wave = 16 co x 16 x outputs for one y row:
//   acc(co, x) = sum_{tap} Wt_tap(16co x 32ci) x im2col_tap(32ci x 16x)
// K per tap = 32 = one WMMA step -> 9 accumulating WMMAs per tile.
// x-halo handled by per-lane zero B-fragments, y-halo by wave-uniform skip.
// C-layout columns are consecutive x -> coalesced f32 stores, fused
// bias + residual.
// ---------------------------------------------------------------------------
__global__ __launch_bounds__(128) void conv_wmma_kernel(
    const bf16_t* __restrict__ AP,   // (B, 4096, 32) bf16 attention output
    const bf16_t* __restrict__ Wt,   // (9, 384, 32) bf16
    const float* __restrict__ bias,  // (384)
    const float* __restrict__ Xin,   // (B, 384, 64, 64)
    float* __restrict__ Out)         // (B, 384, 64, 64)
{
    const int lane = threadIdx.x & 31;
    const int wave = threadIdx.x >> 5;
    const int w    = blockIdx.x * 4 + wave;   // NB*24*64*4 waves
    const int xt   = w & 3;
    const int y    = (w >> 2) & 63;
    const int cg   = w >> 8;                  // b*24 + cot
    const int b    = cg / 24;
    const int cot  = cg - b * 24;
    const int x0   = xt << 4;
    const int co0  = cot << 4;

    const int half = lane >> 4;
    const int l16  = lane & 15;

    v8f acc = {};
    #pragma unroll
    for (int dy = 0; dy < 3; ++dy) {
        int yy = y + dy - 1;
        if ((unsigned)yy < 64u) {             // wave-uniform
            #pragma unroll
            for (int dx = 0; dx < 3; ++dx) {
                int tap = dy * 3 + dx;
                // A-fragment: weights, row m = co, ci 8*half..+7 then 16+8*half..+7
                Frag16 a;
                const bf16_t* wp = Wt + ((size_t)tap * CIN + co0 + l16) * HD + half * 8;
                a.u[0] = *(const uint4*)(wp);
                a.u[1] = *(const uint4*)(wp + 16);
                // B-fragment: im2col, column n = x position, ci 8*half..+7 then 16+8*half..+7
                Frag16 bb;
                int ix = x0 + l16 + dx - 1;
                if ((unsigned)ix < 64u) {
                    const bf16_t* pp = AP + ((size_t)(b * NPOS) + yy * 64 + ix) * HD + half * 8;
                    bb.u[0] = *(const uint4*)(pp);
                    bb.u[1] = *(const uint4*)(pp + 16);
                } else {
                    bb.u[0] = make_uint4(0u, 0u, 0u, 0u);
                    bb.u[1] = make_uint4(0u, 0u, 0u, 0u);
                }
                acc = __builtin_amdgcn_wmma_f32_16x16x32_bf16(false, a.v, false, bb.v,
                                                              (short)0, acc, false, false);
            }
        }
    }

    #pragma unroll
    for (int v = 0; v < 8; ++v) {
        int co = co0 + v + half * 8;
        size_t idx = ((size_t)(b * CIN + co)) * NPOS + y * 64 + x0 + l16;
        Out[idx] = acc[v] + bias[co] + Xin[idx];
    }
}

// ---------------------------------------------------------------------------
extern "C" void kernel_launch(void* const* d_in, const int* in_sizes, int n_in,
                              void* d_out, int out_size, void* d_ws, size_t ws_size,
                              hipStream_t stream)
{
    const float* x    = (const float*)d_in[0];
    const float* xe   = (const float*)d_in[1];
    const float* Wq   = (const float*)d_in[2];
    const float* Wkv  = (const float*)d_in[3];
    const float* Wout = (const float*)d_in[4];
    const float* bout = (const float*)d_in[5];
    float* out = (float*)d_out;

    bf16_t* q     = (bf16_t*)d_ws;                          // 1 MB
    bf16_t* k     = q  + (size_t)NB * NPOS * HD;            // 1 MB
    bf16_t* vT    = k  + (size_t)NB * NPOS * HD;            // 1 MB
    bf16_t* attnP = vT + (size_t)NB * NPOS * HD;            // 1 MB
    bf16_t* Wt    = attnP + (size_t)NB * NPOS * HD;         // 216 KB

    const float scale = 0.17677669529663689f;               // 1/sqrt(32)

    repack_w_kernel<<<(9 * CIN * HD) / 256, 256, 0, stream>>>(Wout, Wt);
    proj_q_kernel <<<(NB * NPOS * HD) / 256, 256, 0, stream>>>(x, Wq, q, scale);
    proj_kv_kernel<<<(NB * NPOS * 64) / 256, 256, 0, stream>>>(xe, Wkv, k, vT);
    attn_kernel   <<<(NB * 256) / 4, 128, 0, stream>>>(q, k, vT, attnP);
    conv_wmma_kernel<<<(NB * 24 * 64 * 4) / 4, 128, 0, stream>>>(attnP, Wt, bout, x, out);
}
